// ConvMultiHeadAttention_76733885710810
// MI455X (gfx1250) — matrix operand order repaired and checked by reading, not compiled, not run on hardware
//
#include <hip/hip_runtime.h>
#include <hip/hip_bf16.h>
#include <math.h>

typedef __bf16 bf16_t;
typedef __attribute__((ext_vector_type(16))) __bf16         v16bf;
typedef __attribute__((ext_vector_type(8)))  float          v8f;
typedef __attribute__((ext_vector_type(8)))  unsigned short v8us;
typedef __attribute__((ext_vector_type(4)))  unsigned int   v4u;
typedef __attribute__((ext_vector_type(8)))  int            v8i;
typedef __attribute__((ext_vector_type(4)))  int            v4i;

#define N_ATOMS  8192
#define N_BONDS  131072
#define DIM      32

#if __has_builtin(__builtin_amdgcn_tensor_load_to_lds) && __has_builtin(__builtin_amdgcn_s_wait_tensorcnt)
#define HAS_TDM 1
#else
#define HAS_TDM 0
#endif

__device__ __forceinline__ bf16_t f2bf(float f) { return (bf16_t)f; }

// ---------------------------------------------------------------------------
// Prep: convert weights to bf16 (B-operand friendly layouts) and zero the
// segment-sum accumulator. W_edge / b_edge natural order (i*32+j) is already
// N-major for the edge GEMM. W_x / W_h (k-major 32x96) are transposed to
// N-major (96x32).
// ---------------------------------------------------------------------------
__global__ void prep_kernel(const float* __restrict__ We, const float* __restrict__ be,
                            const float* __restrict__ Wx, const float* __restrict__ Wh,
                            bf16_t* __restrict__ Webf, bf16_t* __restrict__ bebf,
                            bf16_t* __restrict__ Wxt,  bf16_t* __restrict__ Wht,
                            float* __restrict__ agg) {
    int i = blockIdx.x * blockDim.x + threadIdx.x;
    const int nWe  = 3 * 64 * 1024;      // 196608
    const int nbe  = 3 * 1024;           // 3072
    const int nWt  = 3 * 96 * 32;        // 9216
    const int nAgg = 3 * N_ATOMS * DIM;  // 786432
    if (i < nWe) Webf[i] = f2bf(We[i]);
    if (i < nbe) bebf[i] = f2bf(be[i]);
    if (i < nWt) {
        int v = i / 3072, rem = i % 3072, n = rem / 32, k = rem % 32;
        Wxt[i] = f2bf(Wx[v * 3072 + k * 96 + n]);
        Wht[i] = f2bf(Wh[v * 3072 + k * 96 + n]);
    }
    if (i < nAgg) agg[i] = 0.0f;
}

// ---------------------------------------------------------------------------
// Edge messages:  msg[e,i] = sum_c sum_j edge_in[e,c]*neigh[e,j]*We[c,i*32+j]
//                          + sum_j be[i*32+j]*neigh[e,j]
// One wave per 16-edge tile. K-chunk per c: A = neigh scaled by edge_in[:,c],
// B = We[c] (already N-major in memory). Bias folds in as a 65th chunk.
// edge_in staging uses the Tensor Data Mover: a padded 2-D tile load for the
// bond block (LDS row stride 33 dwords -> conflict-free scalar broadcasts)
// and a gather-mode load for state[bond_g] with 16-bit row indices.
// ---------------------------------------------------------------------------
__global__ void edge_msg_kernel(const float* __restrict__ atom,
                                const float* __restrict__ bond,
                                const float* __restrict__ state,
                                const long long* __restrict__ pair,
                                const long long* __restrict__ bondg,
                                const bf16_t* __restrict__ Webf,
                                const bf16_t* __restrict__ bebf,
                                float* __restrict__ agg) {
    // per-wave slot: 16 rows x 33 dwords (bond, padded) + same for state
    __shared__ float sE[8 * 1056];
    const int wv   = threadIdx.x >> 5;
    const int lane = threadIdx.x & 31;
    const int v    = blockIdx.y;
    const int tile = blockIdx.x * 8 + wv;
    const int ebase = tile * 16;
    const int hi = lane >> 4;
    const int lm = lane & 15;
    const int M  = lm;
    const int j0 = hi * 8;    // unused (A uses kb, like B)
    const int kb = hi * 16;   // B-layout K offset
    float* slot = &sE[wv * 1056];

#if HAS_TDM
    {
        unsigned lds_bond  = (unsigned)(size_t)(void*)slot;
        unsigned lds_state = lds_bond + 2112u;
        unsigned long long ga_bond  = (unsigned long long)(const void*)(bond + (size_t)ebase * DIM);
        unsigned long long ga_state = (unsigned long long)(const void*)state;
        // hoist the 16 gather row-indices to SGPRs
        int bg = (int)bondg[ebase + (lane & 15)];
        unsigned p[8];
#pragma unroll
        for (int i = 0; i < 8; ++i) {
            unsigned lo = (unsigned)__builtin_amdgcn_readlane(bg, 2 * i) & 0xFFFFu;
            unsigned hw = (unsigned)__builtin_amdgcn_readlane(bg, 2 * i + 1) & 0xFFFFu;
            p[i] = lo | (hw << 16);
        }
        // D# group1: data_size=4B, pad_enable, pad every 32 dwords by 1 dword,
        // tensor_dim0=32, tensor_dim1=16, tile=32x16, dim0_stride=32
        unsigned g1w0 = (2u << 16) | (1u << 20) | (4u << 22);
        v8i g1 = { (int)g1w0, (int)(32u << 16), (int)(16u << 16), (int)(32u << 16),
                   16, 32, 0, 0 };
        v4i zz4 = { 0, 0, 0, 0 };
        v8i zz8 = { 0, 0, 0, 0, 0, 0, 0, 0 };
        // bond tile: linear 2-D load
        v4u g0b = { 1u,
                    (unsigned)__builtin_amdgcn_readfirstlane((int)lds_bond),
                    (unsigned)__builtin_amdgcn_readfirstlane((int)(unsigned)ga_bond),
                    ((unsigned)__builtin_amdgcn_readfirstlane((int)(unsigned)(ga_bond >> 32)) & 0x01FFFFFFu) | (2u << 30) };
        __builtin_amdgcn_tensor_load_to_lds(g0b, g1, zz4, zz4, zz8, 0);
        // state rows: gather mode, 16-bit indices
        v4u g0s = { 1u | (1u << 31),
                    (unsigned)__builtin_amdgcn_readfirstlane((int)lds_state),
                    (unsigned)ga_state,
                    ((unsigned)(ga_state >> 32) & 0x01FFFFFFu) | (2u << 30) };
        v4i g2 = { (int)p[0], (int)p[1], (int)p[2], (int)p[3] };
        v4i g3 = { (int)p[4], (int)p[5], (int)p[6], (int)p[7] };
        __builtin_amdgcn_tensor_load_to_lds(g0s, g1, g2, g3, zz8, 0);
    }
#else
    for (int t = lane; t < 512; t += 32) {
        int eL = t >> 5, c = t & 31;
        slot[eL * 33 + c]       = bond[(size_t)(ebase + eL) * DIM + c];
        slot[528 + eL * 33 + c] = state[(size_t)bondg[ebase + eL] * DIM + c];
    }
#endif

    // Per-lane neighbor features in A-layout slots (overlaps the DMA)
    const long long src = pair[(long long)(ebase + M) * 2 + 1];
    const float* np = atom + (size_t)src * DIM;
    float nf[16];
#pragma unroll
    for (int i = 0; i < 8; ++i) { nf[i] = np[kb + i]; nf[8 + i] = np[kb + 8 + i]; }

#if HAS_TDM
    __builtin_amdgcn_s_wait_tensorcnt(0);
    __asm__ volatile("" ::: "memory");
#endif

    v8f acc0 = {0, 0, 0, 0, 0, 0, 0, 0};
    v8f acc1 = {0, 0, 0, 0, 0, 0, 0, 0};
    const float* erow = slot + M * 33;

    for (int c = 0; c <= 64; ++c) {
        float s = (c < 64) ? erow[(c < 32) ? c : (496 + c)] : 1.0f;
        v16bf a;
#pragma unroll
        for (int i = 0; i < 16; ++i) a[i] = f2bf(nf[i] * s);
        const bf16_t* bp = (c < 64) ? (Webf + (size_t)(v * 64 + c) * 1024)
                                    : (bebf + (size_t)v * 1024);
        if (c < 63) __builtin_prefetch(Webf + (size_t)(v * 64 + c + 1) * 1024 + lane * 32, 0, 1);
        v16bf b0 = *reinterpret_cast<const v16bf*>(bp + (0 * 16 + lm) * 32 + kb);
        v16bf b1 = *reinterpret_cast<const v16bf*>(bp + (1 * 16 + lm) * 32 + kb);
        acc0 = __builtin_amdgcn_wmma_f32_16x16x32_bf16(false, a, false, b0, (short)0, acc0, false, false);
        acc1 = __builtin_amdgcn_wmma_f32_16x16x32_bf16(false, a, false, b1, (short)0, acc1, false, false);
    }

    float* aggv = agg + (size_t)v * N_ATOMS * DIM;
#pragma unroll
    for (int r = 0; r < 8; ++r) {
        int e = ebase + r + 8 * hi;                 // C-layout row
        long long dst = pair[(long long)e * 2 + 0]; // segment target
        atomicAdd(&aggv[(size_t)dst * DIM + lm],      acc0[r]);
        atomicAdd(&aggv[(size_t)dst * DIM + 16 + lm], acc1[r]);
    }
}

// ---------------------------------------------------------------------------
// GRU (STEPS=1): mx = agg@Wx+bx (N=96), mh = h@Wh+bh, gates, h'.
// One wave per 16-atom tile. Stores q/k/v as bf16 rows; for v (head 2) also
// stores the transpose (feature-major) for the attention B-operand.
// ---------------------------------------------------------------------------
__global__ void gru_kernel(const float* __restrict__ atom,
                           const float* __restrict__ agg,
                           const bf16_t* __restrict__ Wxt,
                           const bf16_t* __restrict__ Wht,
                           const float* __restrict__ bx,
                           const float* __restrict__ bh,
                           bf16_t* __restrict__ qkv,
                           bf16_t* __restrict__ vT) {
    const int wv = threadIdx.x >> 5, lane = threadIdx.x & 31;
    const int v    = blockIdx.y;
    const int tile = blockIdx.x * 8 + wv;
    const int abase = tile * 16;
    const int hi = lane >> 4, lm = lane & 15;
    const int M = lm, j0 = hi * 8, kb = hi * 16;

    const float* aggv = agg + (size_t)v * N_ATOMS * DIM;
    v16bf aX, aH;
    {
        const float* pa = aggv + (size_t)(abase + M) * DIM;
        const float* ph = atom + (size_t)(abase + M) * DIM;
#pragma unroll
        for (int i = 0; i < 8; ++i) { aX[i] = f2bf(pa[kb + i]); aX[8 + i] = f2bf(pa[kb + 8 + i]); }
#pragma unroll
        for (int i = 0; i < 8; ++i) { aH[i] = f2bf(ph[kb + i]); aH[8 + i] = f2bf(ph[kb + 8 + i]); }
    }
    const bf16_t* WxtV = Wxt + (size_t)v * 96 * 32;
    const bf16_t* WhtV = Wht + (size_t)v * 96 * 32;
    v8f mx[6], mh[6];
#pragma unroll
    for (int t = 0; t < 6; ++t) {
        v8f z = {0, 0, 0, 0, 0, 0, 0, 0};
        int n = t * 16 + lm;
        v16bf bX = *reinterpret_cast<const v16bf*>(WxtV + n * 32 + kb);
        v16bf bH = *reinterpret_cast<const v16bf*>(WhtV + n * 32 + kb);
        mx[t] = __builtin_amdgcn_wmma_f32_16x16x32_bf16(false, aX, false, bX, (short)0, z, false, false);
        mh[t] = __builtin_amdgcn_wmma_f32_16x16x32_bf16(false, aH, false, bH, (short)0, z, false, false);
        float bxv = bx[v * 96 + n], bhv = bh[v * 96 + n];
#pragma unroll
        for (int r = 0; r < 8; ++r) { mx[t][r] += bxv; mh[t][r] += bhv; }
    }
    bf16_t* q = qkv + (size_t)v * N_ATOMS * DIM;
#pragma unroll
    for (int r = 0; r < 8; ++r) {
        int row = abase + r + 8 * hi;
        float h0 = atom[(size_t)row * DIM + lm];
        float h1 = atom[(size_t)row * DIM + 16 + lm];
        float z0 = 1.0f / (1.0f + __expf(-(mx[0][r] + mh[0][r])));
        float z1 = 1.0f / (1.0f + __expf(-(mx[1][r] + mh[1][r])));
        float r0 = 1.0f / (1.0f + __expf(-(mx[2][r] + mh[2][r])));
        float r1 = 1.0f / (1.0f + __expf(-(mx[3][r] + mh[3][r])));
        float hc0 = tanhf(mx[4][r] + r0 * mh[4][r]);
        float hc1 = tanhf(mx[5][r] + r1 * mh[5][r]);
        float hn0 = z0 * h0 + (1.0f - z0) * hc0;
        float hn1 = z1 * h1 + (1.0f - z1) * hc1;
        q[(size_t)row * DIM + lm]      = f2bf(hn0);
        q[(size_t)row * DIM + 16 + lm] = f2bf(hn1);
        if (v == 2) {
            vT[(size_t)lm * N_ATOMS + row]        = f2bf(hn0);
            vT[(size_t)(16 + lm) * N_ATOMS + row] = f2bf(hn1);
        }
    }
}

// ---------------------------------------------------------------------------
// Attention: out = softmax(q k^T / sqrt(32)) v.  One wave per 16-row q tile,
// two passes over k-chunks of 32. Pass 1: row max (WMMA + shfl_xor reduce).
// Pass 2: recompute scores, exp, C->A relayout through LDS, WMMA against vT.
// ---------------------------------------------------------------------------
__global__ void attn_kernel(const bf16_t* __restrict__ qkv,
                            const bf16_t* __restrict__ vT,
                            float* __restrict__ out) {
    __shared__ unsigned short sP[8][16 * 40];   // P tile per wave, padded stride
    const int wv = threadIdx.x >> 5, lane = threadIdx.x & 31;
    const int tile = blockIdx.x * 8 + wv;
    const int qbase = tile * 16;
    const int hi = lane >> 4, lm = lane & 15;
    const int M = lm, j0 = hi * 8, kb = hi * 16;
    const float scale = 0.17677669529663687f;   // 1/sqrt(32)

    const bf16_t* qmat = qkv;                               // head 0
    const bf16_t* kmat = qkv + (size_t)N_ATOMS * DIM;       // head 1

    v16bf aQ;
    {
        const bf16_t* pq = qmat + (size_t)(qbase + M) * DIM;
#pragma unroll
        for (int i = 0; i < 8; ++i) { aQ[i] = pq[kb + i]; aQ[8 + i] = pq[kb + 8 + i]; }
    }

    float m8[8];
#pragma unroll
    for (int r = 0; r < 8; ++r) m8[r] = -3.0e38f;

    // ---- pass 1: raw-score row max ----
    for (int ch = 0; ch < N_ATOMS / 32; ++ch) {
#pragma unroll
        for (int t = 0; t < 2; ++t) {
            int kr = ch * 32 + t * 16 + lm;
            v16bf bK = *reinterpret_cast<const v16bf*>(kmat + (size_t)kr * DIM + kb);
            v8f z = {0, 0, 0, 0, 0, 0, 0, 0};
            v8f s = __builtin_amdgcn_wmma_f32_16x16x32_bf16(false, aQ, false, bK, (short)0, z, false, false);
#pragma unroll
            for (int r = 0; r < 8; ++r) m8[r] = fmaxf(m8[r], s[r]);
        }
    }
#pragma unroll
    for (int d = 1; d < 16; d <<= 1)
#pragma unroll
        for (int r = 0; r < 8; ++r) m8[r] = fmaxf(m8[r], __shfl_xor(m8[r], d, 16));

    // ---- pass 2: exp-weighted accumulation ----
    v8f o0 = {0, 0, 0, 0, 0, 0, 0, 0};
    v8f o1 = {0, 0, 0, 0, 0, 0, 0, 0};
    float s8[8];
#pragma unroll
    for (int r = 0; r < 8; ++r) s8[r] = 0.0f;

    for (int ch = 0; ch < N_ATOMS / 32; ++ch) {
#pragma unroll
        for (int t = 0; t < 2; ++t) {
            int kr = ch * 32 + t * 16 + lm;
            v16bf bK = *reinterpret_cast<const v16bf*>(kmat + (size_t)kr * DIM + kb);
            v8f z = {0, 0, 0, 0, 0, 0, 0, 0};
            v8f s = __builtin_amdgcn_wmma_f32_16x16x32_bf16(false, aQ, false, bK, (short)0, z, false, false);
#pragma unroll
            for (int r = 0; r < 8; ++r) {
                float p = __expf((s[r] - m8[r]) * scale);
                s8[r] += p;
                sP[wv][(r + 8 * hi) * 40 + t * 16 + lm] = __builtin_bit_cast(unsigned short, f2bf(p));
            }
        }
        // read P tile back in A layout (wave-private LDS slot, no barrier)
        v16bf aP;
        {
            const unsigned short* base = &sP[wv][M * 40];
            v8us lo  = *reinterpret_cast<const v8us*>(base + kb);
            v8us hi8 = *reinterpret_cast<const v8us*>(base + kb + 8);
#pragma unroll
            for (int i = 0; i < 8; ++i) {
                aP[i]     = __builtin_bit_cast(bf16_t, (unsigned short)lo[i]);
                aP[8 + i] = __builtin_bit_cast(bf16_t, (unsigned short)hi8[i]);
            }
        }
        v16bf bV0 = *reinterpret_cast<const v16bf*>(vT + (size_t)(0 * 16 + lm) * N_ATOMS + ch * 32 + kb);
        v16bf bV1 = *reinterpret_cast<const v16bf*>(vT + (size_t)(1 * 16 + lm) * N_ATOMS + ch * 32 + kb);
        o0 = __builtin_amdgcn_wmma_f32_16x16x32_bf16(false, aP, false, bV0, (short)0, o0, false, false);
        o1 = __builtin_amdgcn_wmma_f32_16x16x32_bf16(false, aP, false, bV1, (short)0, o1, false, false);
    }

#pragma unroll
    for (int d = 1; d < 16; d <<= 1)
#pragma unroll
        for (int r = 0; r < 8; ++r) s8[r] += __shfl_xor(s8[r], d, 16);

#pragma unroll
    for (int r = 0; r < 8; ++r) {
        int row = qbase + r + 8 * hi;
        float inv = 1.0f / s8[r];
        out[(size_t)row * DIM + lm]      = o0[r] * inv;
        out[(size_t)row * DIM + 16 + lm] = o1[r] * inv;
    }
}

// ---------------------------------------------------------------------------
extern "C" void kernel_launch(void* const* d_in, const int* in_sizes, int n_in,
                              void* d_out, int out_size, void* d_ws, size_t ws_size,
                              hipStream_t stream) {
    const float* atom  = (const float*)d_in[0];
    const float* bond  = (const float*)d_in[1];
    const float* state = (const float*)d_in[2];
    const float* We    = (const float*)d_in[3];
    const float* be    = (const float*)d_in[4];
    const float* Wx    = (const float*)d_in[5];
    const float* Wh    = (const float*)d_in[6];
    const float* bx    = (const float*)d_in[7];
    const float* bh    = (const float*)d_in[8];
    const long long* pair  = (const long long*)d_in[9];
    const long long* bondg = (const long long*)d_in[11];
    (void)in_sizes; (void)n_in; (void)out_size; (void)ws_size;

    char* ws = (char*)d_ws;
    size_t off = 0;
    auto carve = [&](size_t bytes) -> void* {
        void* p = ws + off;
        off = (off + bytes + 255) & ~(size_t)255;
        return p;
    };
    bf16_t* Webf = (bf16_t*)carve((size_t)3 * 64 * 1024 * 2);
    bf16_t* bebf = (bf16_t*)carve((size_t)3 * 1024 * 2);
    bf16_t* Wxt  = (bf16_t*)carve((size_t)3 * 96 * 32 * 2);
    bf16_t* Wht  = (bf16_t*)carve((size_t)3 * 96 * 32 * 2);
    float*  agg  = (float*) carve((size_t)3 * N_ATOMS * DIM * 4);
    bf16_t* qkv  = (bf16_t*)carve((size_t)3 * N_ATOMS * DIM * 2);
    bf16_t* vTm  = (bf16_t*)carve((size_t)N_ATOMS * DIM * 2);

    hipLaunchKernelGGL(prep_kernel, dim3(3072), dim3(256), 0, stream,
                       We, be, Wx, Wh, Webf, bebf, Wxt, Wht, agg);
    hipLaunchKernelGGL(edge_msg_kernel, dim3(N_BONDS / 16 / 8, 3), dim3(256), 0, stream,
                       atom, bond, state, pair, bondg, Webf, bebf, agg);
    hipLaunchKernelGGL(gru_kernel, dim3(N_ATOMS / 16 / 8, 3), dim3(256), 0, stream,
                       atom, agg, Wxt, Wht, bx, bh, qkv, vTm);
    hipLaunchKernelGGL(attn_kernel, dim3(N_ATOMS / 16 / 8), dim3(256), 0, stream,
                       qkv, vTm, (float*)d_out);
}
